// VLSLSTM_13580686590609
// MI455X (gfx1250) — matrix operand order repaired and checked by reading, not compiled, not run on hardware
//
#include <hip/hip_runtime.h>
#include <hip/hip_bf16.h>
#include <cstdint>

// ---------------------------------------------------------------------------
// Problem constants (from reference)
// ---------------------------------------------------------------------------
#define BB   64
#define TTF  512
#define TAR  256
#define HH   512
#define LL   2
#define G4H  (4 * HH)          // 2048

// ---------------------------------------------------------------------------
// Types for CDNA5 WMMA (wave32)
// ---------------------------------------------------------------------------
typedef __bf16  bf16_t;
typedef __attribute__((ext_vector_type(8)))  __bf16 v8bf;
typedef __attribute__((ext_vector_type(16))) __bf16 v16bf;
typedef __attribute__((ext_vector_type(8)))  float  v8f;
typedef __attribute__((ext_vector_type(4)))  int    v4i;

// CDNA5 async global->LDS path (ASYNCcnt-tracked), guarded per-toolchain.
#if defined(__gfx1250__) && __has_builtin(__builtin_amdgcn_global_load_async_to_lds_b128) && __has_builtin(__builtin_amdgcn_s_wait_asynccnt)
#define USE_ASYNC_LDS 1
#else
#define USE_ASYNC_LDS 0
#endif

// round-to-nearest-even f32 -> bf16 (bit pattern)
static __device__ __forceinline__ unsigned short f32_to_bf16_bits(float f) {
    unsigned u = __float_as_uint(f);
    unsigned rnd = 0x7FFFu + ((u >> 16) & 1u);
    return (unsigned short)((u + rnd) >> 16);
}

// ---------------------------------------------------------------------------
// Fragment loaders following CDNA5 ISA §7.12.2 VGPR layouts
// A: 16x32 bf16. Lane L (L<16) = row L; VGPR0..3 hold k=0..7 (pairs),
//    VGPR4..7 hold k=16..23. Lanes 16..31: same rows, k offset +8.
// Source here is the block's LDS stage (row-major, ld = 512).
// ---------------------------------------------------------------------------
static __device__ __forceinline__ v16bf load_a_frag_lds(const unsigned short* A,
                                                        int kbase, int lane) {
    int row = lane & 15;
    int kh  = (lane >> 4) * 8;          // 0 or 8
    const __bf16* p = (const __bf16*)(A + row * HH + kbase + kh);
    v8bf lo = *(const v8bf*)(p);        // k = off .. off+7
    v8bf hi = *(const v8bf*)(p + 16);   // k = off+16 .. off+23
    return __builtin_shufflevector(lo, hi, 0, 1, 2, 3, 4, 5, 6, 7,
                                           8, 9, 10, 11, 12, 13, 14, 15);
}

// B: 32x16 bf16 (Bmat = W^T). Lane L: column n = L&15;
// lanes 0-15 hold k=0..7,16..23, lanes 16-31 hold k=8..15,24..31 (matches A).
// Bmat[k][n] = W[n][k], so each lane reads k=kh..kh+7 and kh+16..kh+23 from row n of W.
static __device__ __forceinline__ v16bf load_b_frag(const unsigned short* W,
                                                    int n0, int kbase, int lane) {
    int col = n0 + (lane & 15);
    int kh  = (lane >> 4) * 8;          // 0 or 8
    const __bf16* p = (const __bf16*)(W + col * HH + kbase + kh);
    v8bf lo = *(const v8bf*)(p);
    v8bf hi = *(const v8bf*)(p + 16);
    return __builtin_shufflevector(lo, hi, 0, 1, 2, 3, 4, 5, 6, 7,
                                           8, 9, 10, 11, 12, 13, 14, 15);
}

// 16-byte global -> LDS copy (async on CDNA5, fallback through VGPRs)
static __device__ __forceinline__ void copy16_g2l(const unsigned short* gsrc,
                                                  unsigned short* ldst) {
#if USE_ASYNC_LDS
    // two-step casts: addrspacecast first, then pointee-type reinterpret
    const __attribute__((address_space(1))) unsigned short* g1 =
        (const __attribute__((address_space(1))) unsigned short*)gsrc;
    __attribute__((address_space(3))) unsigned short* l3 =
        (__attribute__((address_space(3))) unsigned short*)ldst;
    __builtin_amdgcn_global_load_async_to_lds_b128(
        (__attribute__((address_space(1))) v4i*)g1,
        (__attribute__((address_space(3))) v4i*)l3,
        /*offset=*/0, /*cpol=*/0);
#else
    *(v8bf*)ldst = *(const v8bf*)gsrc;
#endif
}

// ---------------------------------------------------------------------------
// Per-step gate GEMM:  g(64 x 2048) = A1(64xK1)@W1^T + A2(64xK2)@W2^T + bias
// A1 = layer input (bf16), A2 = recurrent h state (bf16), W row-major [n][k].
// Grid: 16 blocks = 4 M-tiles x 4 N-groups. Each block stages its 16x1024
// A-slice (32KB) in LDS via ASYNCcnt-tracked loads; 8 waves x 4 N-tiles each.
// ---------------------------------------------------------------------------
__global__ __launch_bounds__(256) void vlslstm_gate_gemm(
        const unsigned short* __restrict__ A1, int lda1,
        const unsigned short* __restrict__ W1,
        const unsigned short* __restrict__ A2, int lda2,
        const unsigned short* __restrict__ W2,
        const float* __restrict__ bias,
        float* __restrict__ g) {
    __shared__ __align__(16) unsigned short sA[2 * 16 * HH];   // 32 KB

    const int lane = threadIdx.x & 31;
    const int wave = threadIdx.x >> 5;
    const int mt = blockIdx.x & 3;            // 4 M-tiles (B=64)
    const int ng = blockIdx.x >> 2;           // 4 N-groups
    const int m0 = mt * 16;
    const int nt0 = ng * 32 + wave * 4;       // first of this wave's 4 N-tiles

    // ---- stage A1/A2 16-row slices into LDS (16B chunks, all 256 threads) ----
    {
        const unsigned short* A1g = A1 + m0 * lda1;
        const unsigned short* A2g = A2 + m0 * lda2;
        const int tid = threadIdx.x;
#pragma unroll
        for (int i = 0; i < 4; ++i) {
            int c   = tid + i * 256;          // 0..1023: 16B chunk id
            int row = c >> 6;                 // 64 chunks per 512-elem row
            int col = (c & 63) * 8;           // element offset (8 bf16 / 16B)
            copy16_g2l(A1g + row * lda1 + col, sA + row * HH + col);
            copy16_g2l(A2g + row * lda2 + col, sA + 16 * HH + row * HH + col);
        }
#if USE_ASYNC_LDS
        __builtin_amdgcn_s_wait_asynccnt(0);
#endif
        __syncthreads();
    }

    v8f acc0 = {0.f, 0.f, 0.f, 0.f, 0.f, 0.f, 0.f, 0.f};
    v8f acc1 = acc0, acc2 = acc0, acc3 = acc0;

    // ---- x-part: A1(slice in LDS) @ W1^T ----
#pragma unroll 2
    for (int kb = 0; kb < HH; kb += 32) {
        v16bf a = load_a_frag_lds(sA, kb, lane);
        v16bf b0 = load_b_frag(W1, (nt0 + 0) * 16, kb, lane);
        v16bf b1 = load_b_frag(W1, (nt0 + 1) * 16, kb, lane);
        v16bf b2 = load_b_frag(W1, (nt0 + 2) * 16, kb, lane);
        v16bf b3 = load_b_frag(W1, (nt0 + 3) * 16, kb, lane);
        acc0 = __builtin_amdgcn_wmma_f32_16x16x32_bf16(false, a, false, b0, (short)0, acc0, false, false);
        acc1 = __builtin_amdgcn_wmma_f32_16x16x32_bf16(false, a, false, b1, (short)0, acc1, false, false);
        acc2 = __builtin_amdgcn_wmma_f32_16x16x32_bf16(false, a, false, b2, (short)0, acc2, false, false);
        acc3 = __builtin_amdgcn_wmma_f32_16x16x32_bf16(false, a, false, b3, (short)0, acc3, false, false);
    }
    // ---- h-part: A2(slice in LDS) @ W2^T ----
#pragma unroll 2
    for (int kb = 0; kb < HH; kb += 32) {
        v16bf a = load_a_frag_lds(sA + 16 * HH, kb, lane);
        v16bf b0 = load_b_frag(W2, (nt0 + 0) * 16, kb, lane);
        v16bf b1 = load_b_frag(W2, (nt0 + 1) * 16, kb, lane);
        v16bf b2 = load_b_frag(W2, (nt0 + 2) * 16, kb, lane);
        v16bf b3 = load_b_frag(W2, (nt0 + 3) * 16, kb, lane);
        acc0 = __builtin_amdgcn_wmma_f32_16x16x32_bf16(false, a, false, b0, (short)0, acc0, false, false);
        acc1 = __builtin_amdgcn_wmma_f32_16x16x32_bf16(false, a, false, b1, (short)0, acc1, false, false);
        acc2 = __builtin_amdgcn_wmma_f32_16x16x32_bf16(false, a, false, b2, (short)0, acc2, false, false);
        acc3 = __builtin_amdgcn_wmma_f32_16x16x32_bf16(false, a, false, b3, (short)0, acc3, false, false);
    }

    // C/D layout: VGPR v, lane<16 -> M=v, N=lane ; lane>=16 -> M=v+8, N=lane-16
    const int mrow = m0 + (lane >> 4) * 8;
    v8f accs[4] = {acc0, acc1, acc2, acc3};
#pragma unroll
    for (int q = 0; q < 4; ++q) {
        const int col = (nt0 + q) * 16 + (lane & 15);
        const float bv = bias[col];
#pragma unroll
        for (int v = 0; v < 8; ++v) {
            g[(mrow + v) * G4H + col] = accs[q][v] + bv;
        }
    }
}

// ---------------------------------------------------------------------------
// Pointwise LSTM cell epilogue: gates -> (h,c) with length masking.
// feed_masked=1 : feed buffer gets masked output (TF inter-layer / AR x_{t+1})
// feed_masked=0 : feed buffer gets raw h_n (AR inter-layer input)
// ---------------------------------------------------------------------------
__global__ __launch_bounds__(256) void vlslstm_pointwise(
        const float* __restrict__ g,
        float* __restrict__ h, float* __restrict__ c,
        unsigned short* __restrict__ h_bf,
        unsigned short* __restrict__ feed, int feed_masked,
        float* __restrict__ out, int out_row_stride,
        const int* __restrict__ lengths, int t) {
    int idx = blockIdx.x * blockDim.x + threadIdx.x;   // B*H = 32768
    if (idx >= BB * HH) return;
    int b = idx >> 9;          // / 512
    int j = idx & (HH - 1);    // % 512

    const float* gr = g + b * G4H;
    float gi = gr[j];
    float gf = gr[HH + j];
    float gc = gr[2 * HH + j];
    float go = gr[3 * HH + j];

    float i = 1.0f / (1.0f + __expf(-gi));
    float f = 1.0f / (1.0f + __expf(-gf));
    float o = 1.0f / (1.0f + __expf(-go));
    float gt = tanhf(gc);

    float cn = f * c[idx] + i * gt;
    float hn = o * tanhf(cn);

    bool m = t < lengths[b];
    float hs = m ? hn : h[idx];
    float cs = m ? cn : c[idx];
    h[idx] = hs;
    c[idx] = cs;
    h_bf[idx] = f32_to_bf16_bits(hs);

    float mo = m ? hn : 0.0f;
    if (feed) feed[idx] = f32_to_bf16_bits(feed_masked ? mo : hn);
    if (out)  out[b * out_row_stride + t * HH + j] = mo;
}

// ---------------------------------------------------------------------------
// fp32 -> bf16 bulk conversion
// ---------------------------------------------------------------------------
__global__ __launch_bounds__(256) void vlslstm_cvt_bf16(
        const float* __restrict__ src, unsigned short* __restrict__ dst, int n) {
    int i = blockIdx.x * blockDim.x + threadIdx.x;
    if (i < n) dst[i] = f32_to_bf16_bits(src[i]);
}

// ---------------------------------------------------------------------------
// Host-side orchestration
// ---------------------------------------------------------------------------
extern "C" void kernel_launch(void* const* d_in, const int* in_sizes, int n_in,
                              void* d_out, int out_size, void* d_ws, size_t ws_size,
                              hipStream_t stream) {
    (void)in_sizes; (void)n_in; (void)out_size; (void)ws_size;

    const float* x      = (const float*)d_in[0];   // (B, T_TF, H)
    const float* W_ih   = (const float*)d_in[1];   // (L, 4H, H)
    const float* W_hh   = (const float*)d_in[2];   // (L, 4H, H)
    const float* bias   = (const float*)d_in[3];   // (L, 4H)
    const int*   len_tf = (const int*)d_in[4];     // (B,)
    const int*   len_ar = (const int*)d_in[5];     // (B,)

    float* out_tf = (float*)d_out;                         // (B, T_TF, H)
    float* out_ar = (float*)d_out + (size_t)BB * TTF * HH; // (B, T_AR, H)

    // ---- workspace carve-up (all 256B aligned) ----
    char* ws = (char*)d_ws;
    size_t off = 0;
    auto alloc = [&](size_t bytes) -> char* {
        char* p = ws + off;
        off += (bytes + 255) & ~size_t(255);
        return p;
    };
    const size_t nW = (size_t)LL * G4H * HH;        // 2,097,152 per weight array
    unsigned short* Wih_bf = (unsigned short*)alloc(nW * 2);
    unsigned short* Whh_bf = (unsigned short*)alloc(nW * 2);
    unsigned short* X_bf   = (unsigned short*)alloc((size_t)BB * TTF * HH * 2);
    float*          h_st   = (float*)alloc((size_t)LL * BB * HH * 4);
    float*          c_st   = (float*)alloc((size_t)LL * BB * HH * 4);
    unsigned short* h_bf   = (unsigned short*)alloc((size_t)LL * BB * HH * 2);
    unsigned short* a1_bf  = (unsigned short*)alloc((size_t)BB * HH * 2);
    unsigned short* arb0   = (unsigned short*)alloc((size_t)BB * HH * 2);
    unsigned short* arb1   = (unsigned short*)alloc((size_t)BB * HH * 2);
    float*          gbuf   = (float*)alloc((size_t)BB * G4H * 4);

    // ---- one-time conversions (deterministic, inside capture) ----
    {
        int n = (int)nW;
        vlslstm_cvt_bf16<<<(n + 255) / 256, 256, 0, stream>>>(W_ih, Wih_bf, n);
        vlslstm_cvt_bf16<<<(n + 255) / 256, 256, 0, stream>>>(W_hh, Whh_bf, n);
        int nx = BB * TTF * HH;
        vlslstm_cvt_bf16<<<(nx + 255) / 256, 256, 0, stream>>>(x, X_bf, nx);
    }

    // ---- zero initial state ----
    (void)hipMemsetAsync(h_st, 0, (size_t)LL * BB * HH * 4, stream);
    (void)hipMemsetAsync(c_st, 0, (size_t)LL * BB * HH * 4, stream);
    (void)hipMemsetAsync(h_bf, 0, (size_t)LL * BB * HH * 2, stream);

    const dim3 gemm_grid(16), gemm_blk(256);
    const dim3 pw_grid((BB * HH + 255) / 256), pw_blk(256);

    const int ldX = TTF * HH;   // row (batch) stride of x at fixed t

    // =========================== teacher-forced phase ========================
    for (int t = 0; t < TTF; ++t) {
        // layer 0: input = x[:, t, :]
        vlslstm_gate_gemm<<<gemm_grid, gemm_blk, 0, stream>>>(
            X_bf + (size_t)t * HH, ldX, Wih_bf,
            h_bf, HH, Whh_bf,
            bias, gbuf);
        vlslstm_pointwise<<<pw_grid, pw_blk, 0, stream>>>(
            gbuf, h_st, c_st, h_bf,
            a1_bf, /*feed_masked=*/1,
            /*out=*/nullptr, 0, len_tf, t);

        // layer 1: input = masked h0
        vlslstm_gate_gemm<<<gemm_grid, gemm_blk, 0, stream>>>(
            a1_bf, HH, Wih_bf + nW / LL,
            h_bf + BB * HH, HH, Whh_bf + nW / LL,
            bias + G4H, gbuf);
        vlslstm_pointwise<<<pw_grid, pw_blk, 0, stream>>>(
            gbuf, h_st + BB * HH, c_st + BB * HH, h_bf + BB * HH,
            /*feed=*/nullptr, 0,
            out_tf, TTF * HH, len_tf, t);
    }

    // =========================== autoregressive phase ========================
    for (int t = 0; t < TAR; ++t) {
        const unsigned short* arIn =
            (t == 0) ? (h_bf + BB * HH)                 // x0 = final TF h of layer 1
                     : ((t - 1) & 1 ? arb1 : arb0);     // previous masked output
        unsigned short* arOut = (t & 1) ? arb1 : arb0;

        // layer 0
        vlslstm_gate_gemm<<<gemm_grid, gemm_blk, 0, stream>>>(
            arIn, HH, Wih_bf,
            h_bf, HH, Whh_bf,
            bias, gbuf);
        vlslstm_pointwise<<<pw_grid, pw_blk, 0, stream>>>(
            gbuf, h_st, c_st, h_bf,
            a1_bf, /*feed_masked=*/0,       // inter-layer feed is UNmasked h_n
            /*out=*/nullptr, 0, len_ar, t);

        // layer 1
        vlslstm_gate_gemm<<<gemm_grid, gemm_blk, 0, stream>>>(
            a1_bf, HH, Wih_bf + nW / LL,
            h_bf + BB * HH, HH, Whh_bf + nW / LL,
            bias + G4H, gbuf);
        vlslstm_pointwise<<<pw_grid, pw_blk, 0, stream>>>(
            gbuf, h_st + BB * HH, c_st + BB * HH, h_bf + BB * HH,
            arOut, /*feed_masked=*/1,       // next-step input & output are masked
            out_ar, TAR * HH, len_ar, t);
    }
}
